// Gate_55697135894809
// MI455X (gfx1250) — matrix operand order repaired and checked by reading, not compiled, not run on hardware
//
#include <hip/hip_runtime.h>
#include <stdint.h>

typedef __attribute__((ext_vector_type(16))) _Float16 v16h;
typedef __attribute__((ext_vector_type(8)))  float    v8f;
typedef __attribute__((ext_vector_type(4)))  float    f32x4;
typedef __attribute__((ext_vector_type(4)))  uint32_t u32x4;

#define D_DIM 7168
#define E_DIM 256
#define ROUTE_SCALE 2.5f
#define NEG_BIG (-3.0e38f)

union HPair { _Float16 h[2]; uint32_t u; };
union BFrag { u32x4 q[2]; v16h h; };

// Non-temporal vector load: x is streamed exactly once -> keep WGP$/L2 for W.
__device__ __forceinline__ f32x4 nt_load4(const float* p) {
    return __builtin_nontemporal_load((const f32x4*)p);
}

// ---------------------------------------------------------------------------
// Kernel 0: pack W [E,D] fp32 -> f16 pairs in WMMA-B-fragment order.
// dword index = ((k2>>3)*E + e)*8 + (k2&7), where k2 = K/2 (one dword = 2 K).
// A lane's B-fragment (8 dwords) is two 16B pieces 8KB apart (K and K+16).
// ---------------------------------------------------------------------------
__global__ void pack_w_kernel(const float* __restrict__ W,
                              uint32_t* __restrict__ bpack) {
    int t   = blockIdx.x * blockDim.x + threadIdx.x;   // over (D/2)*E dwords
    int blk = t >> 11;                                  // k2 block (k2>>3)
    int rem = t & 2047;
    int e   = rem >> 3;
    int k2  = (blk << 3) | (rem & 7);
    const float* src = W + (size_t)e * D_DIM + 2 * k2;
    HPair p;
    p.h[0] = (_Float16)src[0];
    p.h[1] = (_Float16)src[1];
    bpack[t] = p.u;                                     // coalesced write
}

// ---------------------------------------------------------------------------
// Kernel 1: scores = sigmoid(x @ W^T) via v_wmma_f32_16x16x32_f16.
// Per-wave tile: M=32 tokens x N=128 experts (16 C-tiles = 128 acc VGPRs).
// Each B fragment feeds 2 WMMAs -> per-iter cache traffic 12KB/wave
// (2 fp32 A-frags + 8 f16 B-frags) vs 18KB for the skinny M=16xN=256 tile.
// Block = 4 waves = 2 token-groups x 2 expert-halves (64 tokens/block).
// ---------------------------------------------------------------------------
__global__ void __launch_bounds__(128, 1)
gemm_sig_kernel(const float* __restrict__ x,
                const uint32_t* __restrict__ bpack,
                float* __restrict__ scores) {
    const int lane  = threadIdx.x & 31;
    const int wave  = threadIdx.x >> 5;        // 0..3
    const int l15   = lane & 15;
    const int lhi   = lane >> 4;               // 0: lanes 0-15, 1: lanes 16-31
    const int t0    = (blockIdx.x * 2 + (wave >> 1)) * 32;   // 32-token group
    const int ebase = (wave & 1) * 128;                      // expert half

    const float* xrow0 = x + (size_t)(t0 + l15) * D_DIM;        // M-tile 0
    const float* xrow1 = xrow0 + (size_t)16 * D_DIM;            // M-tile 1

    v8f acc[16];                                // [n] m=0, [8+n] m=1
#pragma unroll
    for (int n = 0; n < 16; ++n)
#pragma unroll
        for (int i = 0; i < 8; ++i) acc[n][i] = 0.0f;

    for (int kc = 0; kc < D_DIM; kc += 32) {
        // ---- A fragments (16x32 f16 each), ISA layout:
        // lanes 0-15: V0-3 = K0..7, V4-7 = K16..23
        // lanes16-31: V0-3 = K8..15, V4-7 = K24..31
        const int k0 = kc + (lhi << 3);
        f32x4 xa0 = nt_load4(xrow0 + k0);
        f32x4 xb0 = nt_load4(xrow0 + k0 + 4);
        f32x4 xc0 = nt_load4(xrow0 + k0 + 16);
        f32x4 xd0 = nt_load4(xrow0 + k0 + 20);
        f32x4 xa1 = nt_load4(xrow1 + k0);
        f32x4 xb1 = nt_load4(xrow1 + k0 + 4);
        f32x4 xc1 = nt_load4(xrow1 + k0 + 16);
        f32x4 xd1 = nt_load4(xrow1 + k0 + 20);
        v16h a0, a1;
#pragma unroll
        for (int i = 0; i < 4; ++i) {
            a0[i]      = (_Float16)xa0[i];
            a0[4 + i]  = (_Float16)xb0[i];
            a0[8 + i]  = (_Float16)xc0[i];
            a0[12 + i] = (_Float16)xd0[i];
            a1[i]      = (_Float16)xa1[i];
            a1[4 + i]  = (_Float16)xb1[i];
            a1[8 + i]  = (_Float16)xc1[i];
            a1[12 + i] = (_Float16)xd1[i];
        }

        // ---- B fragments: lanes 0-15 K0..7,K16..23; lanes 16-31 K8..15,K24..31.
        const uint32_t* bbase =
            bpack + (size_t)((kc >> 4) * E_DIM + ebase + l15) * 8 + lhi * 4;
#pragma unroll
        for (int n = 0; n < 8; ++n) {
            BFrag bf;
            const u32x4* bp = (const u32x4*)(bbase + n * 128);  // e += 16
            bf.q[0] = bp[0];
            bf.q[1] = bp[E_DIM * 2];
            acc[n] = __builtin_amdgcn_wmma_f32_16x16x32_f16(
                false, a0, false, bf.h, (short)0, acc[n], false, false);
            acc[8 + n] = __builtin_amdgcn_wmma_f32_16x16x32_f16(
                false, a1, false, bf.h, (short)0, acc[8 + n], false, false);
        }
    }

    // ---- Epilogue: sigmoid + store. C layout: VGPR v, lanes0-15 -> M=v,
    // lanes16-31 -> M=v+8; N = lane&15 within tile n.
#pragma unroll
    for (int m = 0; m < 2; ++m) {
        const int trow = t0 + m * 16 + (lhi << 3);
#pragma unroll
        for (int n = 0; n < 8; ++n)
#pragma unroll
            for (int v = 0; v < 8; ++v) {
                float s = 1.0f / (1.0f + __expf(-acc[m * 8 + n][v]));
                scores[(size_t)(trow + v) * E_DIM + ebase + n * 16 + l15] = s;
            }
    }
}

// ---------------------------------------------------------------------------
// Kernel 2: top-8 per token (wave per token). Selection on scores+bias,
// weights from unbiased scores, renormalized, x2.5. Tie -> lower index
// (matches jax.lax.top_k).
// ---------------------------------------------------------------------------
__global__ void topk_kernel(const float* __restrict__ scores,
                            const float* __restrict__ bias,
                            float* __restrict__ outw,
                            int* __restrict__ outi, int T) {
    const int lane = threadIdx.x & 31;
    const int wave = threadIdx.x >> 5;
    const int t = blockIdx.x * 8 + wave;
    if (t >= T) return;

    const float* row = scores + (size_t)t * E_DIM;
    float s[8], b[8];
#pragma unroll
    for (int j = 0; j < 8; ++j) {
        int e = j * 32 + lane;          // coalesced per j
        s[j] = row[e];
        b[j] = s[j] + bias[e];
    }

    float wsum = 0.0f;
    float myw = 0.0f;
    int   myi = 0;

#pragma unroll
    for (int r = 0; r < 8; ++r) {
        // local argmax over this lane's 8 experts (ascending j => lowest idx)
        float bv = NEG_BIG; int bi = E_DIM; float bs = 0.0f;
#pragma unroll
        for (int j = 0; j < 8; ++j) {
            int e = j * 32 + lane;
            if (b[j] > bv || (b[j] == bv && e < bi)) { bv = b[j]; bi = e; bs = s[j]; }
        }
        // butterfly reduction: all 32 lanes converge on the winner
        for (int off = 16; off > 0; off >>= 1) {
            float ov = __shfl_xor(bv, off, 32);
            int   oi = __shfl_xor(bi, off, 32);
            float os = __shfl_xor(bs, off, 32);
            if (ov > bv || (ov == bv && oi < bi)) { bv = ov; bi = oi; bs = os; }
        }
        wsum += bs;
        if (lane == r) { myw = bs; myi = bi; }   // r is a compile-time constant
        if ((bi & 31) == lane) b[bi >> 5] = NEG_BIG;  // knock out winner
    }

    const float inv = ROUTE_SCALE / wsum;
    if (lane < 8) {
        outw[(size_t)t * 8 + lane] = myw * inv;
        outi[(size_t)t * 8 + lane] = myi;
    }
}

// ---------------------------------------------------------------------------
extern "C" void kernel_launch(void* const* d_in, const int* in_sizes, int n_in,
                              void* d_out, int out_size, void* d_ws, size_t ws_size,
                              hipStream_t stream) {
    const float* x    = (const float*)d_in[0];   // [T, D]
    const float* W    = (const float*)d_in[1];   // [E, D]
    const float* bias = (const float*)d_in[2];   // [E]

    const long E = in_sizes[2];             // 256
    const long D = in_sizes[1] / E;         // 7168
    const long T = in_sizes[0] / D;         // 16384

    float*    scores = (float*)d_ws;                                   // T*E f32
    uint32_t* bpack  = (uint32_t*)((char*)d_ws + (size_t)T * E * sizeof(float));

    // Phase 0: pack W -> f16 B-fragment layout (L2-resident, 3.7 MB)
    pack_w_kernel<<<(int)((D / 2 * E) / 256), 256, 0, stream>>>(W, bpack);

    // Phase 1: WMMA GEMM + sigmoid. 4 waves/block, 64 tokens/block.
    gemm_sig_kernel<<<(int)(T / 64), 128, 0, stream>>>(x, bpack, scores);

    // Phase 2: top-8 + renorm. 8 tokens per 256-thread block.
    float* outw = (float*)d_out;
    int*   outi = (int*)((float*)d_out + (size_t)T * 8);
    topk_kernel<<<(int)((T + 7) / 8), 256, 0, stream>>>(scores, bias, outw, outi, (int)T);
}
